// MultiHeadSelfAttentionModule_49855980372563
// MI455X (gfx1250) — matrix operand order repaired and checked by reading, not compiled, not run on hardware
//
#include <hip/hip_runtime.h>
#include <hip/hip_bf16.h>

typedef __attribute__((ext_vector_type(16))) _Float16 v16h;
typedef __attribute__((ext_vector_type(8)))  _Float16 h8v;
typedef __attribute__((ext_vector_type(8)))  float    v8f;

#define Bn 4
#define Sn 2048
#define Dn 512
#define Hn 8
#define DKn 64
#define Nn (Bn * Sn)   // 8192 tokens

// ---------------- fragment loaders (contiguous k: lane half h holds k 16h..16h+15) ----------------

// A-matrix 16x32 f16: lane holds row (lane%16); halves 0..15 = K[2*kbase..2*kbase+15],
// kbase = 8*(lane/16) (contiguous, matching make_fragB).
__device__ __forceinline__ v16h make_fragA(const _Float16* row, int kbase) {
    h8v lo = *(const h8v*)(row + 2 * kbase);
    h8v hi = *(const h8v*)(row + 2 * kbase + 8);
    v16h a;
#pragma unroll
    for (int j = 0; j < 8; ++j) { a[j] = lo[j]; a[j + 8] = hi[j]; }
    return a;
}

// B-matrix 32x16 f16: lane holds column (lane%16); 16 contiguous K values starting
// at 16*(lane/16). Caller passes the pointer already offset for this lane.
__device__ __forceinline__ v16h make_fragB(const _Float16* colk) {
    h8v lo = *(const h8v*)(colk);
    h8v hi = *(const h8v*)(colk + 8);
    v16h b;
#pragma unroll
    for (int j = 0; j < 8; ++j) { b[j] = lo[j]; b[j + 8] = hi[j]; }
    return b;
}

__device__ __forceinline__ v8f wmma_f16(v16h a, v16h b, v8f c) {
    return __builtin_amdgcn_wmma_f32_16x16x32_f16(false, a, false, b, (short)0, c,
                                                  false, false);
}

// ---------------- kernel 1: LayerNorm (fp32 in -> f16 out) ----------------
__global__ void ln_kernel(const float* __restrict__ x, const float* __restrict__ gamma,
                          const float* __restrict__ beta, _Float16* __restrict__ xh) {
    int row = blockIdx.x;                 // 8192 rows
    int tid = threadIdx.x;                // 256 threads, 2 elems each
    const float* xr = x + (size_t)row * Dn;
    float2 v = ((const float2*)xr)[tid];
    float s  = v.x + v.y;
    float ss = v.x * v.x + v.y * v.y;
#pragma unroll
    for (int m = 16; m >= 1; m >>= 1) {
        s  += __shfl_xor(s, m, 32);
        ss += __shfl_xor(ss, m, 32);
    }
    __shared__ float red[16];
    int w = tid >> 5, lane = tid & 31;
    if (lane == 0) { red[w] = s; red[8 + w] = ss; }
    __syncthreads();
    float tot = 0.f, tot2 = 0.f;
#pragma unroll
    for (int i = 0; i < 8; ++i) { tot += red[i]; tot2 += red[8 + i]; }
    float mu   = tot * (1.f / (float)Dn);
    float var  = tot2 * (1.f / (float)Dn) - mu * mu;
    float rstd = rsqrtf(var + 1e-5f);
    float g0 = gamma[2 * tid], g1 = gamma[2 * tid + 1];
    float b0 = beta[2 * tid],  b1 = beta[2 * tid + 1];
    _Float16* xo = xh + (size_t)row * Dn;
    xo[2 * tid]     = (_Float16)((v.x - mu) * rstd * g0 + b0);
    xo[2 * tid + 1] = (_Float16)((v.y - mu) * rstd * g1 + b1);
}

// ---------------- kernel 2: fp32 -> f16 conversion ----------------
__global__ void cvt_kernel(const float* __restrict__ src, _Float16* __restrict__ dst, int n) {
    int i = blockIdx.x * blockDim.x + threadIdx.x;
    if (i < n) dst[i] = (_Float16)src[i];
}

// ---------------- kernel 3: QKV projection (WMMA, 32x64 tile per wave) ----------------
// Wave computes M=32 x N=64; N-block == one head (DK=64).
__global__ void qkv_gemm(const _Float16* __restrict__ xh,
                         const _Float16* __restrict__ Wqh, const _Float16* __restrict__ Wkh,
                         const _Float16* __restrict__ Wvh,
                         const float* __restrict__ bq, const float* __restrict__ bk,
                         const float* __restrict__ bv,
                         _Float16* __restrict__ Qh, _Float16* __restrict__ Kh,
                         _Float16* __restrict__ Vth) {
    int wid  = blockIdx.x * 8 + (threadIdx.x >> 5);
    int lane = threadIdx.x & 31;
    int proj = wid >> 11;          // 256 * 8 = 2048 tiles per projection
    int rem  = wid & 2047;
    int mt   = rem >> 3;           // 0..255 token tiles of 32
    int nt   = rem & 7;            // 0..7 heads (64 features each)
    const _Float16* Wp = (proj == 0) ? Wqh : (proj == 1) ? Wkh : Wvh;
    const float*    bp = (proj == 0) ? bq  : (proj == 1) ? bk  : bv;

    int lm = lane & 15, hs = lane >> 4, kbase = hs * 8;
    const _Float16* arow0 = xh + (size_t)(mt * 32 + lm) * Dn;
    const _Float16* arow1 = arow0 + (size_t)16 * Dn;
    const _Float16* brow[4];
#pragma unroll
    for (int j = 0; j < 4; ++j)
        brow[j] = Wp + (size_t)(nt * 64 + j * 16 + lm) * Dn + hs * 16;

    v8f zero = {};
    v8f c[2][4];
#pragma unroll
    for (int i = 0; i < 2; ++i)
#pragma unroll
        for (int j = 0; j < 4; ++j) c[i][j] = zero;

#pragma unroll 2
    for (int kk = 0; kk < Dn; kk += 32) {
        v16h a0 = make_fragA(arow0 + kk, kbase);
        v16h a1 = make_fragA(arow1 + kk, kbase);
#pragma unroll
        for (int j = 0; j < 4; ++j) {
            v16h b = make_fragB(brow[j] + kk);
            c[0][j] = wmma_f16(a0, b, c[0][j]);
            c[1][j] = wmma_f16(a1, b, c[1][j]);
        }
    }

    int hh = nt;                          // head index
    size_t bhbase = 0;                    // computed per row below
#pragma unroll
    for (int j = 0; j < 4; ++j) {
        int dk = j * 16 + lm;
        float bias = bp[nt * 64 + dk];
#pragma unroll
        for (int i = 0; i < 2; ++i) {
#pragma unroll
            for (int r = 0; r < 8; ++r) {
                int t = mt * 32 + i * 16 + r + hs * 8;
                int bb = t >> 11, s = t & (Sn - 1);
                _Float16 hv = (_Float16)(c[i][j][r] + bias);
                size_t bh = (size_t)(bb * Hn + hh);
                if (proj == 0)      Qh[(bh * Sn + s) * DKn + dk] = hv;
                else if (proj == 1) Kh[(bh * Sn + s) * DKn + dk] = hv;
                else                Vth[(bh * DKn + dk) * Sn + s] = hv;
            }
        }
    }
    (void)bhbase;
}

// ---------------- kernel 4: flash attention (WMMA, 32 queries per wave) ----------------
// Block = 4 waves of one (b,h); each wave owns two 16-query tiles sharing K/V frags.
__global__ void attn_kernel(const _Float16* __restrict__ Qh, const _Float16* __restrict__ Kh,
                            const _Float16* __restrict__ Vth, const int* __restrict__ lengths,
                            _Float16* __restrict__ attnh) {
    __shared__ _Float16 sP[4][2][16][40];   // 80B row stride -> 16B-aligned ds_load_b128
    int w = threadIdx.x >> 5, lane = threadIdx.x & 31;
    int lm = lane & 15, hs = lane >> 4, kbase = hs * 8;

    int bid = blockIdx.x;                  // 512 blocks
    int b   = bid >> 7;                    // same b across block -> uniform trip count
    int rem = bid & 127;
    int h   = rem >> 4;
    int qg  = (rem & 15) * 4 + w;          // 0..63 groups of 32 queries
    int len = lengths[b];

    size_t bh = (size_t)(b * Hn + h);
    const _Float16* Qb = Qh + (bh * Sn + qg * 32) * DKn;
    const _Float16* Kb = Kh + bh * Sn * DKn;
    const _Float16* Vb = Vth + bh * DKn * Sn;

    // Q fragments: qf[qi][chunk], qi = query tile 0/1, chunk = dk 0..31 / 32..63
    v16h qf[2][2];
#pragma unroll
    for (int qi = 0; qi < 2; ++qi) {
        const _Float16* qrow = Qb + (size_t)(qi * 16 + lm) * DKn;
        qf[qi][0] = make_fragA(qrow, kbase);
        qf[qi][1] = make_fragA(qrow + 32, kbase);
    }

    float mrow[2][8], lrow[2][8];
    v8f o[2][4];
    v8f zero = {};
#pragma unroll
    for (int qi = 0; qi < 2; ++qi) {
#pragma unroll
        for (int r = 0; r < 8; ++r) { mrow[qi][r] = -__builtin_inff(); lrow[qi][r] = 0.f; }
#pragma unroll
        for (int d = 0; d < 4; ++d) o[qi][d] = zero;
    }

    int nChunks = (len + 31) >> 5;
    for (int ck = 0; ck < nChunks; ++ck) {
        int key0 = ck * 32;
        const _Float16* kp0 = Kb + (size_t)(key0 + lm) * DKn + hs * 16;
        const _Float16* kp1 = kp0 + (size_t)16 * DKn;

        // prefetch next chunk of K and V into cache
        __builtin_prefetch(kp0 + (size_t)32 * DKn, 0, 0);
        __builtin_prefetch(kp1 + (size_t)32 * DKn, 0, 0);
        __builtin_prefetch(Vb + (size_t)lm * Sn + key0 + 32, 0, 0);

        // K fragments shared by both query tiles
        v16h kb00 = make_fragB(kp0);
        v16h kb01 = make_fragB(kp0 + 32);
        v16h kb10 = make_fragB(kp1);
        v16h kb11 = make_fragB(kp1 + 32);

        v8f sc[2][2];
#pragma unroll
        for (int qi = 0; qi < 2; ++qi) {
            sc[qi][0] = wmma_f16(qf[qi][0], kb00, zero);
            sc[qi][0] = wmma_f16(qf[qi][1], kb01, sc[qi][0]);
            sc[qi][1] = wmma_f16(qf[qi][0], kb10, zero);
            sc[qi][1] = wmma_f16(qf[qi][1], kb11, sc[qi][1]);
        }

        bool ok0 = (key0 + lm) < len;
        bool ok1 = (key0 + 16 + lm) < len;

        __syncthreads();                   // WAR vs previous iteration's sP reads
#pragma unroll
        for (int qi = 0; qi < 2; ++qi) {
#pragma unroll
            for (int r = 0; r < 8; ++r) {
                float a0 = ok0 ? sc[qi][0][r] * 0.125f : -__builtin_inff();
                float a1 = ok1 ? sc[qi][1][r] * 0.125f : -__builtin_inff();
                float rm = fmaxf(a0, a1);
                rm = fmaxf(rm, __shfl_xor(rm, 1, 32));
                rm = fmaxf(rm, __shfl_xor(rm, 2, 32));
                rm = fmaxf(rm, __shfl_xor(rm, 4, 32));
                rm = fmaxf(rm, __shfl_xor(rm, 8, 32));
                float mnew  = fmaxf(mrow[qi][r], rm);
                float alpha = __expf(mrow[qi][r] - mnew);
                float p0 = __expf(a0 - mnew);
                float p1 = __expf(a1 - mnew);
                float ps = p0 + p1;
                ps += __shfl_xor(ps, 1, 32);
                ps += __shfl_xor(ps, 2, 32);
                ps += __shfl_xor(ps, 4, 32);
                ps += __shfl_xor(ps, 8, 32);
                lrow[qi][r] = lrow[qi][r] * alpha + ps;
                mrow[qi][r] = mnew;
                o[qi][0][r] *= alpha; o[qi][1][r] *= alpha;
                o[qi][2][r] *= alpha; o[qi][3][r] *= alpha;
                int rr = r + hs * 8;       // C-layout row
                sP[w][qi][rr][lm]      = (_Float16)p0;
                sP[w][qi][rr][16 + lm] = (_Float16)p1;
            }
        }
        __syncthreads();

        v16h pa0 = make_fragA(&sP[w][0][lm][0], kbase);   // repack C->A layout
        v16h pa1 = make_fragA(&sP[w][1][lm][0], kbase);
#pragma unroll
        for (int d = 0; d < 4; ++d) {
            const _Float16* vr = Vb + (size_t)(d * 16 + lm) * Sn + key0 + hs * 16;
            v16h vbF = make_fragB(vr);                    // shared by both q tiles
            o[0][d] = wmma_f16(pa0, vbF, o[0][d]);
            o[1][d] = wmma_f16(pa1, vbF, o[1][d]);
        }
    }

#pragma unroll
    for (int qi = 0; qi < 2; ++qi) {
#pragma unroll
        for (int r = 0; r < 8; ++r) {
            float inv = 1.f / lrow[qi][r];
            int row = qg * 32 + qi * 16 + r + hs * 8;
            size_t base = ((size_t)b * Sn + row) * Dn + h * DKn + lm;
            attnh[base]      = (_Float16)(o[qi][0][r] * inv);
            attnh[base + 16] = (_Float16)(o[qi][1][r] * inv);
            attnh[base + 32] = (_Float16)(o[qi][2][r] * inv);
            attnh[base + 48] = (_Float16)(o[qi][3][r] * inv);
        }
    }
}

// ---------------- kernel 5: output projection (WMMA, 32x64 tile, fp32 out) ----------------
__global__ void out_gemm(const _Float16* __restrict__ attnh, const _Float16* __restrict__ Woh,
                         const float* __restrict__ bo, float* __restrict__ out) {
    int wid  = blockIdx.x * 8 + (threadIdx.x >> 5);
    int lane = threadIdx.x & 31;
    int mt = wid >> 3;     // 0..255
    int nt = wid & 7;      // 0..7
    int lm = lane & 15, hs = lane >> 4, kbase = hs * 8;
    const _Float16* arow0 = attnh + (size_t)(mt * 32 + lm) * Dn;
    const _Float16* arow1 = arow0 + (size_t)16 * Dn;
    const _Float16* brow[4];
#pragma unroll
    for (int j = 0; j < 4; ++j)
        brow[j] = Woh + (size_t)(nt * 64 + j * 16 + lm) * Dn + hs * 16;

    v8f zero = {};
    v8f c[2][4];
#pragma unroll
    for (int i = 0; i < 2; ++i)
#pragma unroll
        for (int j = 0; j < 4; ++j) c[i][j] = zero;

#pragma unroll 2
    for (int kk = 0; kk < Dn; kk += 32) {
        v16h a0 = make_fragA(arow0 + kk, kbase);
        v16h a1 = make_fragA(arow1 + kk, kbase);
#pragma unroll
        for (int j = 0; j < 4; ++j) {
            v16h b = make_fragB(brow[j] + kk);
            c[0][j] = wmma_f16(a0, b, c[0][j]);
            c[1][j] = wmma_f16(a1, b, c[1][j]);
        }
    }

#pragma unroll
    for (int j = 0; j < 4; ++j) {
        int col = nt * 64 + j * 16 + lm;
        float bias = bo[col];
#pragma unroll
        for (int i = 0; i < 2; ++i) {
#pragma unroll
            for (int r = 0; r < 8; ++r) {
                int t = mt * 32 + i * 16 + r + hs * 8;
                out[(size_t)t * Dn + col] = c[i][j][r] + bias;
            }
        }
    }
}

// ---------------- launcher ----------------
extern "C" void kernel_launch(void* const* d_in, const int* in_sizes, int n_in,
                              void* d_out, int out_size, void* d_ws, size_t ws_size,
                              hipStream_t stream) {
    const float* x        = (const float*)d_in[0];
    const int*   lengths  = (const int*)d_in[1];
    // d_in[2] = pos_embed (unused by reference)
    const float* ln_gamma = (const float*)d_in[3];
    const float* ln_beta  = (const float*)d_in[4];
    const float* Wq = (const float*)d_in[5];  const float* bq = (const float*)d_in[6];
    const float* Wk = (const float*)d_in[7];  const float* bk = (const float*)d_in[8];
    const float* Wv = (const float*)d_in[9];  const float* bv = (const float*)d_in[10];
    const float* Wo = (const float*)d_in[11]; const float* bo = (const float*)d_in[12];
    float* out = (float*)d_out;

    char* ws = (char*)d_ws;
    const size_t MB = 1u << 20;
    _Float16* xh    = (_Float16*)(ws);                 // 8 MiB (8192*512 f16)
    _Float16* Qh    = (_Float16*)(ws + 8 * MB);        // 8 MiB
    _Float16* Kh    = (_Float16*)(ws + 16 * MB);       // 8 MiB
    _Float16* Vth   = (_Float16*)(ws + 24 * MB);       // 8 MiB (transposed V)
    _Float16* attnh = (_Float16*)(ws + 32 * MB);       // 8 MiB
    _Float16* Wqh   = (_Float16*)(ws + 40 * MB);       // 512 KiB each
    _Float16* Wkh   = Wqh + (size_t)Dn * Dn;
    _Float16* Wvh   = Wkh + (size_t)Dn * Dn;
    _Float16* Woh   = Wvh + (size_t)Dn * Dn;

    // 1. LayerNorm -> f16
    ln_kernel<<<Nn, 256, 0, stream>>>(x, ln_gamma, ln_beta, xh);

    // 2. weight conversion
    const int nw = Dn * Dn;
    cvt_kernel<<<nw / 256, 256, 0, stream>>>(Wq, Wqh, nw);
    cvt_kernel<<<nw / 256, 256, 0, stream>>>(Wk, Wkh, nw);
    cvt_kernel<<<nw / 256, 256, 0, stream>>>(Wv, Wvh, nw);
    cvt_kernel<<<nw / 256, 256, 0, stream>>>(Wo, Woh, nw);

    // 3. QKV projection: 3 * 256 * 8 tiles (32x64 each), 8 waves/block
    qkv_gemm<<<768, 256, 0, stream>>>(xh, Wqh, Wkh, Wvh, bq, bk, bv, Qh, Kh, Vth);

    // 4. flash attention: 4 (B) * 8 (H) * 16 (32-query groups / 4 waves) blocks
    attn_kernel<<<512, 128, 0, stream>>>(Qh, Kh, Vth, lengths, attnh);

    // 5. output projection: 256 * 8 tiles (32x64 each), 8 waves/block
    out_gemm<<<256, 256, 0, stream>>>(attnh, Woh, bo, out);
}
